// TriangleAttentionBase_22479858827754
// MI455X (gfx1250) — matrix operand order repaired and checked by reading, not compiled, not run on hardware
//
#include <hip/hip_runtime.h>
#include <hip/hip_bf16.h>

// ---------------------------------------------------------------------------
// Types for CDNA5 WMMA
// ---------------------------------------------------------------------------
typedef __attribute__((ext_vector_type(16))) __bf16        v16bf;
typedef __attribute__((ext_vector_type(2)))  __bf16        v2bf;
typedef __attribute__((ext_vector_type(8)))  float         v8f;
typedef __attribute__((ext_vector_type(4)))  float         v4f;
typedef __attribute__((ext_vector_type(4)))  unsigned int  v4u;

union FragU { v4u u[2]; v16bf b; };

__device__ __forceinline__ unsigned short f2bf(float f) {
  union { float f; unsigned u; } v; v.f = f;
  unsigned r = v.u + 0x7FFFu + ((v.u >> 16) & 1u);  // round-to-nearest-even
  return (unsigned short)(r >> 16);
}

// Packed f32x2 -> bf16x2 (v_cvt_pk_bf16_f32 when available)
__device__ __forceinline__ unsigned pack_bf16(float a, float b) {
#if __has_builtin(__builtin_amdgcn_cvt_pk_bf16_f32)
  union { v2bf v; unsigned u; } c;
  c.v = __builtin_amdgcn_cvt_pk_bf16_f32(a, b);
  return c.u;
#else
  return (unsigned)f2bf(a) | ((unsigned)f2bf(b) << 16);
#endif
}

__device__ __forceinline__ v8f zero8() {
  v8f z;
#pragma unroll
  for (int j = 0; j < 8; ++j) z[j] = 0.0f;
  return z;
}

__device__ __forceinline__ v8f wmma_bf16(v16bf a, v16bf b, v8f c) {
  // D = A(16x32 bf16) * B(32x16 bf16) + C(16x16 f32)
  return __builtin_amdgcn_wmma_f32_16x16x32_bf16(false, a, false, b, (short)0, c,
                                                 false, false);
}

// A-operand 16x32 bf16 fragment from row-major [M][K] (ISA 7.12.2):
// lane holds row (lane&15); VGPR0-3 = cols col0+half*8.., VGPR4-7 = col0+16+half*8..
__device__ __forceinline__ v16bf ldfragA(const unsigned short* base, int row,
                                         int rowElems, int col0, int lrow, int half) {
  const unsigned short* p =
      base + (size_t)(row + lrow) * rowElems + col0 + half * 8;
  FragU f;
  f.u[0] = *(const v4u*)p;
  f.u[1] = *(const v4u*)(p + 16);
  return f.b;
}

// B-operand 32x16 bf16 fragment from row-major B^T tile [N][K] (ISA 7.12.4):
// lane holds column n=(lane&15); VGPR0-3 = K col0+half*8.., VGPR4-7 = col0+16+half*8..
// (same K order as ldfragA).
__device__ __forceinline__ v16bf ldfragB(const unsigned short* base, int row,
                                         int rowElems, int col0, int lrow, int half) {
  const unsigned short* p =
      base + (size_t)(row + lrow) * rowElems + col0 + half * 8;
  FragU f;
  f.u[0] = *(const v4u*)p;
  f.u[1] = *(const v4u*)(p + 16);
  return f.b;
}

// ---------------------------------------------------------------------------
// Constants: N=256 pair side, C=128 channels, D=32 head dim, H=4 heads
// ---------------------------------------------------------------------------
#define NN    256
#define CC    128
#define NPAIR (NN * NN)

// Workspace layout (bytes)
#define WS_XN     0            // bf16 [65536][128]  = 16 MB
#define WS_BIAS   16777216     // f32  [4][256 k][256 q] = 1 MB (TRANSPOSED)
#define WS_WQKVT  17825792     // bf16 [384][128]
#define WS_WGT    17924096     // bf16 [128][128]
#define WS_WOUTT  17956864     // bf16 [128][128]
#define WS_WBM    17989632     // f32  [128][4]
#define WS_BGP    17991680     // f32  [128]

// LDS layout (bytes) for attn kernel
#define L_Q   0                // bf16 [256][128]
#define L_K   65536            // bf16 [256][128]
#define L_VT  131072           // bf16 [4][32][256]   (V transposed per head)
#define L_O   196608           // bf16 [256][128]     (gated attention output)
#define L_P   262144           // bf16 [8 waves][16][32] P-transpose scratch
#define LDS_TOTAL 270336

// ---------------------------------------------------------------------------
// Kernel 1: repack weights -> bf16, transposed, head-major (hd = h*32 + d)
// original last-dim index for (d,h) is d*H + h = d*4 + h
// ---------------------------------------------------------------------------
__global__ void __launch_bounds__(256) prep_weights_kernel(
    const float* __restrict__ wqkv,   // [128][384]
    const float* __restrict__ wb,     // [128][128]
    const float* __restrict__ wg,     // [128][128]
    const float* __restrict__ bg,     // [128]
    const float* __restrict__ wout,   // [128][128]
    unsigned short* __restrict__ wqkvT,  // [384][128] bf16 (rows m*128+h*32+d)
    unsigned short* __restrict__ wgT,    // [128][128] bf16 (rows h*32+d)
    unsigned short* __restrict__ woutT,  // [128][128] bf16 (rows cout, cols h*32+d)
    float* __restrict__ wbm,             // [128][4]  mean_d wb
    float* __restrict__ bgp)             // [128] b_g head-major
{
  int tid = blockIdx.x * blockDim.x + threadIdx.x;
  int nth = gridDim.x * blockDim.x;

  for (int idx = tid; idx < 384 * 128; idx += nth) {
    int r = idx >> 7, c = idx & 127;
    int m = r >> 7, hd = r & 127, h = hd >> 5, d = hd & 31;
    wqkvT[idx] = f2bf(wqkv[c * 384 + m * 128 + d * 4 + h]);
  }
  for (int idx = tid; idx < 128 * 128; idx += nth) {
    int hd = idx >> 7, c = idx & 127, h = hd >> 5, d = hd & 31;
    wgT[idx] = f2bf(wg[c * 128 + d * 4 + h]);
  }
  for (int idx = tid; idx < 128 * 128; idx += nth) {
    int co = idx >> 7, hd = idx & 127, h = hd >> 5, d = hd & 31;
    woutT[idx] = f2bf(wout[(d * 4 + h) * 128 + co]);
  }
  for (int idx = tid; idx < 128 * 4; idx += nth) {
    int c = idx >> 2, h = idx & 3;
    float s = 0.0f;
    for (int d = 0; d < 32; ++d) s += wb[c * 128 + d * 4 + h];
    wbm[idx] = s * (1.0f / 32.0f);
  }
  for (int idx = tid; idx < 128; idx += nth) {
    int h = idx >> 5, d = idx & 31;
    bgp[idx] = bg[d * 4 + h];
  }
}

// ---------------------------------------------------------------------------
// Kernel 2: LayerNorm -> bf16 xn, plus pair bias map (stored TRANSPOSED as
// biasT[h][k][q] so the attention kernel can vector-load it along q).
// One wave per pair position (4 channels per lane).
// ---------------------------------------------------------------------------
__global__ void __launch_bounds__(256) ln_bias_kernel(
    const float* __restrict__ x,     // [65536][128]
    const float* __restrict__ lnw, const float* __restrict__ lnb,
    const float* __restrict__ wbm,   // [128][4]
    unsigned short* __restrict__ xn, // [65536][128] bf16
    float* __restrict__ biasT)       // [4][256][256]  (h, k, q)
{
  int wid = threadIdx.x >> 5;
  int lane = threadIdx.x & 31;
  int pos = blockIdx.x * 8 + wid;
  int c = lane * 4;

  v4f v = __builtin_nontemporal_load((const v4f*)(x + (size_t)pos * CC + c));
  float s  = v[0] + v[1] + v[2] + v[3];
  float sq = v[0] * v[0] + v[1] * v[1] + v[2] * v[2] + v[3] * v[3];
#pragma unroll
  for (int m = 1; m < 32; m <<= 1) {
    s  += __shfl_xor(s, m, 32);
    sq += __shfl_xor(sq, m, 32);
  }
  float mu  = s * (1.0f / 128.0f);
  float var = sq * (1.0f / 128.0f) - mu * mu;
  float rs  = rsqrtf(var + 1e-5f);

  v4f w = *(const v4f*)(lnw + c);
  v4f b = *(const v4f*)(lnb + c);
  float xv[4];
#pragma unroll
  for (int e = 0; e < 4; ++e) xv[e] = (v[e] - mu) * rs * w[e] + b[e];

  uint2 pk;
  pk.x = pack_bf16(xv[0], xv[1]);
  pk.y = pack_bf16(xv[2], xv[3]);
  *(uint2*)(xn + (size_t)pos * CC + c) = pk;

  float acc[4] = {0.f, 0.f, 0.f, 0.f};
#pragma unroll
  for (int e = 0; e < 4; ++e) {
    const float* wm = wbm + (c + e) * 4;
#pragma unroll
    for (int h = 0; h < 4; ++h) acc[h] += xv[e] * wm[h];
  }
#pragma unroll
  for (int h = 0; h < 4; ++h)
#pragma unroll
    for (int m = 1; m < 32; m <<= 1) acc[h] += __shfl_xor(acc[h], m, 32);
  if (lane == 0) {
    int q = pos >> 8, k = pos & 255;       // pair (q=i1, k=i2)
#pragma unroll
    for (int h = 0; h < 4; ++h) biasT[h * NPAIR + k * 256 + q] = acc[h];
  }
}

// ---------------------------------------------------------------------------
// Kernel 3: fused triangle attention, one workgroup (8 waves) per row i.
// ---------------------------------------------------------------------------
__global__ void __launch_bounds__(256) attn_kernel(
    const unsigned short* __restrict__ xn,    // bf16 [65536][128]
    const float* __restrict__ biasT,          // f32  [4][256][256] (h,k,q)
    const unsigned short* __restrict__ wqkvT, // bf16 [384][128]
    const unsigned short* __restrict__ wgT,   // bf16 [128][128]
    const unsigned short* __restrict__ woutT, // bf16 [128][128]
    const float* __restrict__ bgp,            // f32 [128]
    const float* __restrict__ bout,           // f32 [128]
    float* __restrict__ out)                  // f32 [65536][128]
{
  extern __shared__ char smem[];
  unsigned short* Qs = (unsigned short*)(smem + L_Q);
  unsigned short* Ks = (unsigned short*)(smem + L_K);
  unsigned short* Vt = (unsigned short*)(smem + L_VT);
  unsigned short* Os = (unsigned short*)(smem + L_O);
  unsigned short* Pa = (unsigned short*)(smem + L_P);

  const int i    = blockIdx.x;
  const int wid  = threadIdx.x >> 5;
  const int lane = threadIdx.x & 31;
  const int lrow = lane & 15;
  const int half = lane >> 4;

  const unsigned short* xrow = xn + (size_t)i * NN * CC;

  // Prefetch this block's xn row (64 KB, one 256B line per thread).
  __builtin_prefetch(((const char*)xrow) + threadIdx.x * 256, 0, 3);

  // ---------- Phase 1: Q, K (head-major) and V^T into LDS -----------------
  // 3 outputs x 16 pos-tiles x 8 n-tiles = 384 16x16 tiles, 4 WMMA each.
  for (int t = wid; t < 384; t += 8) {
    int m  = t / 128;          // 0=q, 1=k, 2=v
    int r  = t - m * 128;
    int pt = r >> 3;           // pos tile 0..15
    int nt = r & 7;            // out tile 0..7
    v8f acc = zero8();
#pragma unroll
    for (int kc = 0; kc < 4; ++kc) {
      v16bf a = ldfragA(xrow, pt * 16, CC, kc * 32, lrow, half);
      v16bf b = ldfragB(wqkvT, m * 128 + nt * 16, CC, kc * 32, lrow, half);
      acc = wmma_bf16(a, b, acc);
    }
    if (m < 2) {
      unsigned short* dst = (m == 0) ? Qs : Ks;
#pragma unroll
      for (int j = 0; j < 8; ++j) {
        int M = j + half * 8;
        dst[(pt * 16 + M) * CC + nt * 16 + lrow] = f2bf(acc[j]);
      }
    } else {
      int hd = nt * 16 + lrow;
      int h  = hd >> 5, d = hd & 31;
      v4u pv;
      pv[0] = pack_bf16(acc[0], acc[1]);
      pv[1] = pack_bf16(acc[2], acc[3]);
      pv[2] = pack_bf16(acc[4], acc[5]);
      pv[3] = pack_bf16(acc[6], acc[7]);
      *(v4u*)&Vt[h * 8192 + d * 256 + pt * 16 + half * 8] = pv;
    }
  }
  __syncthreads();

  // ---------- Phase 2: per (head, q-tile) attention -----------------------
  const float scale = 0.17677669529663687f;  // 1/sqrt(32)
  for (int task = wid; task < 64; task += 8) {
    int h  = task >> 4;
    int qt = task & 15;

    v16bf qa = ldfragA(Qs, qt * 16, CC, h * 32, lrow, half);

    // Full logits row: 16 tiles of 16x16 f32 (K=32 -> single WMMA per tile)
    v8f s[16];
#pragma unroll
    for (int kt = 0; kt < 16; ++kt) {
      v16bf kb = ldfragB(Ks, kt * 16, CC, h * 32, lrow, half);
      s[kt] = wmma_bf16(qa, kb, zero8());
    }

    // scale + pair bias: biasT[h][k][q] -> per lane 8 contiguous floats in q,
    // matching C-layout register order (q = qt*16 + half*8 + j).
    const float* bh = biasT + ((size_t)h << 16);
#pragma unroll
    for (int kt = 0; kt < 16; ++kt) {
      const float* brow = bh + (kt * 16 + lrow) * 256 + qt * 16 + half * 8;
      v4f b0 = *(const v4f*)brow;
      v4f b1 = *(const v4f*)(brow + 4);
#pragma unroll
      for (int j = 0; j < 8; ++j)
        s[kt][j] = s[kt][j] * scale + (j < 4 ? b0[j] : b1[j - 4]);
    }

    // softmax over k: reduce over 16 tiles then over 16 lanes within half
    float mx[8], sm[8];
#pragma unroll
    for (int j = 0; j < 8; ++j) {
      mx[j] = s[0][j];
#pragma unroll
      for (int kt = 1; kt < 16; ++kt) mx[j] = fmaxf(mx[j], s[kt][j]);
#pragma unroll
      for (int m = 1; m < 16; m <<= 1) mx[j] = fmaxf(mx[j], __shfl_xor(mx[j], m, 32));
    }
#pragma unroll
    for (int j = 0; j < 8; ++j) {
      sm[j] = 0.0f;
#pragma unroll
      for (int kt = 0; kt < 16; ++kt) {
        s[kt][j] = __expf(s[kt][j] - mx[j]);
        sm[j] += s[kt][j];
      }
#pragma unroll
      for (int m = 1; m < 16; m <<= 1) sm[j] += __shfl_xor(sm[j], m, 32);
      sm[j] = 1.0f / sm[j];
#pragma unroll
      for (int kt = 0; kt < 16; ++kt) s[kt][j] *= sm[j];
    }

    // AV: transpose P 16x32 chunks through per-wave LDS scratch, then WMMA
    unsigned short* ps = Pa + wid * 512;  // [16][32] bf16
    v8f o0 = zero8(), o1 = zero8();
#pragma unroll
    for (int c = 0; c < 8; ++c) {
#pragma unroll
      for (int tt = 0; tt < 2; ++tt)
#pragma unroll
        for (int j = 0; j < 8; ++j) {
          int M = j + half * 8;
          ps[M * 32 + tt * 16 + lrow] = f2bf(s[2 * c + tt][j]);
        }
      asm volatile("s_wait_dscnt 0" ::: "memory");  // LDS store->load, same wave
      v16bf pa  = ldfragA(ps, 0, 32, 0, lrow, half);
      v16bf vb0 = ldfragB(Vt + h * 8192, 0, 256, c * 32, lrow, half);            // d 0..15
      v16bf vb1 = ldfragB(Vt + h * 8192 + 16 * 256, 0, 256, c * 32, lrow, half); // d 16..31
      o0 = wmma_bf16(pa, vb0, o0);
      o1 = wmma_bf16(pa, vb1, o1);
    }

    // gate: g = sigmoid(xn @ Wg_h + b_g), computed with WMMA from global xn
    v8f g0 = zero8(), g1 = zero8();
#pragma unroll
    for (int kc = 0; kc < 4; ++kc) {
      v16bf xa  = ldfragA(xrow, qt * 16, CC, kc * 32, lrow, half);
      v16bf wb0 = ldfragB(wgT, h * 32, CC, kc * 32, lrow, half);
      v16bf wb1 = ldfragB(wgT, h * 32 + 16, CC, kc * 32, lrow, half);
      g0 = wmma_bf16(xa, wb0, g0);
      g1 = wmma_bf16(xa, wb1, g1);
    }
    float bg0 = bgp[h * 32 + lrow];
    float bg1 = bgp[h * 32 + 16 + lrow];
#pragma unroll
    for (int j = 0; j < 8; ++j) {
      int M = j + half * 8;
      float gg0 = 1.0f / (1.0f + __expf(-(g0[j] + bg0)));
      float gg1 = 1.0f / (1.0f + __expf(-(g1[j] + bg1)));
      Os[(qt * 16 + M) * CC + h * 32 + lrow]      = f2bf(o0[j] * gg0);
      Os[(qt * 16 + M) * CC + h * 32 + 16 + lrow] = f2bf(o1[j] * gg1);
    }
  }
  __syncthreads();

  // ---------- Phase 3: output projection z = O @ Wout + b_out -------------
  for (int t = wid; t < 128; t += 8) {
    int pt = t >> 3, nt = t & 7;
    v8f acc = zero8();
#pragma unroll
    for (int kc = 0; kc < 4; ++kc) {
      v16bf a = ldfragA(Os, pt * 16, CC, kc * 32, lrow, half);
      v16bf b = ldfragB(woutT, nt * 16, CC, kc * 32, lrow, half);
      acc = wmma_bf16(a, b, acc);
    }
    float bo = bout[nt * 16 + lrow];
    float* dst = out + ((size_t)i * NN + pt * 16) * CC + nt * 16 + lrow;
#pragma unroll
    for (int j = 0; j < 8; ++j) {
      int M = j + half * 8;
      __builtin_nontemporal_store(acc[j] + bo, dst + (size_t)M * CC);
    }
  }
}

// ---------------------------------------------------------------------------
// Host launcher
// ---------------------------------------------------------------------------
extern "C" void kernel_launch(void* const* d_in, const int* in_sizes, int n_in,
                              void* d_out, int out_size, void* d_ws, size_t ws_size,
                              hipStream_t stream) {
  const float* x     = (const float*)d_in[0];
  const float* ln_w  = (const float*)d_in[1];
  const float* ln_b  = (const float*)d_in[2];
  const float* w_qkv = (const float*)d_in[3];
  const float* w_b   = (const float*)d_in[4];
  const float* w_g   = (const float*)d_in[5];
  const float* b_g   = (const float*)d_in[6];
  const float* w_out = (const float*)d_in[7];
  const float* b_out = (const float*)d_in[8];
  float* out = (float*)d_out;

  char* ws = (char*)d_ws;
  unsigned short* xn    = (unsigned short*)(ws + WS_XN);
  float*          biasm = (float*)(ws + WS_BIAS);
  unsigned short* wqkvT = (unsigned short*)(ws + WS_WQKVT);
  unsigned short* wgT   = (unsigned short*)(ws + WS_WGT);
  unsigned short* woutT = (unsigned short*)(ws + WS_WOUTT);
  float*          wbm   = (float*)(ws + WS_WBM);
  float*          bgp   = (float*)(ws + WS_BGP);

  (void)in_sizes; (void)n_in; (void)out_size; (void)ws_size;

  hipFuncSetAttribute((const void*)attn_kernel,
                      hipFuncAttributeMaxDynamicSharedMemorySize, LDS_TOTAL);

  prep_weights_kernel<<<64, 256, 0, stream>>>(w_qkv, w_b, w_g, b_g, w_out,
                                              wqkvT, wgT, woutT, wbm, bgp);
  ln_bias_kernel<<<NPAIR / 8, 256, 0, stream>>>(x, ln_w, ln_b, wbm, xn, biasm);
  attn_kernel<<<NN, 256, LDS_TOTAL, stream>>>(xn, biasm, wqkvT, wgT, woutT,
                                              bgp, b_out, out);
}
